// CausalSelfAttention_41145786695873
// MI455X (gfx1250) — matrix operand order repaired and checked by reading, not compiled, not run on hardware
//
#include <hip/hip_runtime.h>

// ---------------------------------------------------------------------------
// Problem constants (match reference)
// ---------------------------------------------------------------------------
#define BB     2
#define SS     2048
#define HIDD   2048
#define NHEAD  16
#define NKV    8
#define HD     128          // head dim
#define MROWS  (BB * SS)    // 4096

typedef __bf16 bf16_t;
typedef __attribute__((ext_vector_type(16))) __bf16 v16bf;
typedef __attribute__((ext_vector_type(8)))  float   v8f;

// ---------------------------------------------------------------------------
// WMMA helpers (wave32; A/B k order is this file's own, C/D per CDNA5 ISA)
// A 16x32 bf16 : lane holds row M=lane&15; elements 0..15 = K[2*koff..2*koff+16),
//                koff = (lane>=16)*8 (same k order as B)
// B 32x16 bf16 : lane holds col N=lane&15; elements 0..15 = K[k16..k16+16),
//                k16 = (lane>=16)*16
// C/D 16x16 f32: VGPR r -> row (r + (lane>=16)*8), col = lane&15
// ---------------------------------------------------------------------------
__device__ __forceinline__ v8f wmma_bf16(v16bf a, v16bf b, v8f c) {
  return __builtin_amdgcn_wmma_f32_16x16x32_bf16(false, a, false, b,
                                                 (short)0, c, false, false);
}

__device__ __forceinline__ v16bf load_frag_a(const bf16_t* p) {
  // p = &A[row][k0 + koff]; row-major, 16 contiguous elems K[k0+16h..k0+16h+16)
  v16bf a;
  const int hsh = ((threadIdx.x & 31) >> 4) * 8;
#pragma unroll
  for (int i = 0; i < 8; ++i) { a[i] = p[hsh + i]; a[i + 8] = p[hsh + i + 8]; }
  return a;
}

__device__ __forceinline__ v16bf load_frag_b(const bf16_t* p) {
  // p = &Bt[n][k0 + k16]; 16 contiguous elements (32 bytes)
  v16bf b;
#pragma unroll
  for (int i = 0; i < 16; ++i) b[i] = p[i];
  return b;
}

// ---------------------------------------------------------------------------
// Elementwise conversion kernels
// ---------------------------------------------------------------------------
__global__ void f32_to_bf16_kernel(const float* __restrict__ in,
                                   bf16_t* __restrict__ out, int n) {
  int i = blockIdx.x * blockDim.x + threadIdx.x;
  if (i < n) out[i] = (bf16_t)in[i];
}

// W: [K][N] f32 -> Wt: [N][K] bf16
__global__ void transpose_to_bf16_kernel(const float* __restrict__ W,
                                         bf16_t* __restrict__ Wt, int K, int N) {
  int i = blockIdx.x * blockDim.x + threadIdx.x;
  if (i < K * N) {
    int k = i % K;
    int n = i / K;
    Wt[(size_t)n * K + k] = (bf16_t)W[(size_t)k * N + n];
  }
}

// xq f32 [b][s][h*128+d] -> RoPE -> Qb bf16 [b][h][s][128]
__global__ void rope_q_kernel(const float* __restrict__ xq,
                              const float* __restrict__ cosb,
                              const float* __restrict__ sinb,
                              bf16_t* __restrict__ Qb) {
  int i = blockIdx.x * blockDim.x + threadIdx.x;   // ((b*S+s)*16+h)*64+pair
  if (i >= BB * SS * NHEAD * 64) return;
  int pair = i & 63;
  int h = (i >> 6) & 15;
  int s = (i >> 10) & (SS - 1);
  int b = i >> 21;
  float c  = cosb[s * 64 + pair];
  float sn = sinb[s * 64 + pair];
  const float* q = xq + ((size_t)(b * SS + s)) * HIDD + h * HD + pair * 2;
  float re = q[0], im = q[1];
  bf16_t* o = Qb + (((size_t)(b * NHEAD + h)) * SS + s) * HD + pair * 2;
  o[0] = (bf16_t)(re * c - im * sn);
  o[1] = (bf16_t)(re * sn + im * c);
}

// xk f32 [b][s][kv*128+d] -> RoPE -> Kb bf16 [b][kv][s][128]
__global__ void rope_k_kernel(const float* __restrict__ xk,
                              const float* __restrict__ cosb,
                              const float* __restrict__ sinb,
                              bf16_t* __restrict__ Kb) {
  int i = blockIdx.x * blockDim.x + threadIdx.x;   // ((b*S+s)*8+kv)*64+pair
  if (i >= BB * SS * NKV * 64) return;
  int pair = i & 63;
  int kv = (i >> 6) & 7;
  int s = (i >> 9) & (SS - 1);
  int b = i >> 20;
  float c  = cosb[s * 64 + pair];
  float sn = sinb[s * 64 + pair];
  const float* k = xk + ((size_t)(b * SS + s)) * (NKV * HD) + kv * HD + pair * 2;
  float re = k[0], im = k[1];
  bf16_t* o = Kb + (((size_t)(b * NKV + kv)) * SS + s) * HD + pair * 2;
  o[0] = (bf16_t)(re * c - im * sn);
  o[1] = (bf16_t)(re * sn + im * c);
}

// xv f32 [b][s][kv*128+d] -> Vt bf16 [b][kv][d][s]
__global__ void v_transpose_kernel(const float* __restrict__ xv,
                                   bf16_t* __restrict__ Vt) {
  int i = blockIdx.x * blockDim.x + threadIdx.x;   // ((b*8+kv)*128+d)*2048+s
  if (i >= BB * NKV * HD * SS) return;
  int s = i & (SS - 1);
  int d = (i >> 11) & (HD - 1);
  int kv = (i >> 18) & 7;
  int b = i >> 21;
  Vt[i] = (bf16_t)xv[((size_t)(b * SS + s)) * (NKV * HD) + kv * HD + d];
}

// ---------------------------------------------------------------------------
// Generic WMMA GEMM: C[M][N] f32 = A[M][K] bf16 (row-major) * Bt[N][K] bf16
// Block = 128 threads (4 waves) arranged 2x2; wave tile = 32x64 (2m x 4n WMMA
// tiles -> 8 WMMA per 32-deep k-step); block tile = 64x128.
// Requires M % 64 == 0, N % 128 == 0, K % 32 == 0 (true for all calls here).
// ---------------------------------------------------------------------------
__global__ __launch_bounds__(128) void gemm_bf16_wmma_kernel(
    const bf16_t* __restrict__ A, const bf16_t* __restrict__ Bt,
    float* __restrict__ C, int M, int N, int K) {
  int lane = threadIdx.x & 31;
  int wave = threadIdx.x >> 5;                 // 0..3
  int half = lane >> 4;                        // 0/1
  int l16  = lane & 15;
  int m0 = blockIdx.y * 64 + (wave >> 1) * 32;
  int n0 = blockIdx.x * 128 + (wave & 1) * 64;

  const bf16_t* a0p = A + (size_t)(m0 + l16) * K + half * 8;
  const bf16_t* a1p = A + (size_t)(m0 + 16 + l16) * K + half * 8;
  const bf16_t* bp0 = Bt + (size_t)(n0 + l16) * K + half * 16;
  const bf16_t* bp1 = Bt + (size_t)(n0 + 16 + l16) * K + half * 16;
  const bf16_t* bp2 = Bt + (size_t)(n0 + 32 + l16) * K + half * 16;
  const bf16_t* bp3 = Bt + (size_t)(n0 + 48 + l16) * K + half * 16;

  v8f acc0[4], acc1[4];
#pragma unroll
  for (int j = 0; j < 4; ++j) { acc0[j] = (v8f){}; acc1[j] = (v8f){}; }

  const int PF = 256;  // prefetch distance in elements (512 B, 8 k-steps ahead)
  for (int k = 0; k < K; k += 32) {
    if (k + PF < K) {   // gfx1250 global_prefetch_b8 path (L2 -> WGP)
      __builtin_prefetch(a0p + k + PF, 0, 1);
      __builtin_prefetch(bp0 + k + PF, 0, 1);
      __builtin_prefetch(bp2 + k + PF, 0, 1);
    }
    v16bf a0 = load_frag_a(a0p + k);
    v16bf a1 = load_frag_a(a1p + k);
    v16bf b0 = load_frag_b(bp0 + k);
    v16bf b1 = load_frag_b(bp1 + k);
    v16bf b2 = load_frag_b(bp2 + k);
    v16bf b3 = load_frag_b(bp3 + k);
    acc0[0] = wmma_bf16(a0, b0, acc0[0]);
    acc0[1] = wmma_bf16(a0, b1, acc0[1]);
    acc0[2] = wmma_bf16(a0, b2, acc0[2]);
    acc0[3] = wmma_bf16(a0, b3, acc0[3]);
    acc1[0] = wmma_bf16(a1, b0, acc1[0]);
    acc1[1] = wmma_bf16(a1, b1, acc1[1]);
    acc1[2] = wmma_bf16(a1, b2, acc1[2]);
    acc1[3] = wmma_bf16(a1, b3, acc1[3]);
  }

  int rbase0 = m0 + half * 8;
  int rbase1 = m0 + 16 + half * 8;
#pragma unroll
  for (int j = 0; j < 4; ++j) {
    float* c0 = C + (size_t)rbase0 * N + n0 + j * 16 + l16;
    float* c1 = C + (size_t)rbase1 * N + n0 + j * 16 + l16;
#pragma unroll
    for (int r = 0; r < 8; ++r) {
      c0[(size_t)r * N] = acc0[j][r];
      c1[(size_t)r * N] = acc1[j][r];
    }
  }
}

// ---------------------------------------------------------------------------
// Flash-style attention (no mask, matching reference).
// One wave per 16-query block per (b,h). GQA: kv = h/2.
// Qb [b][h][s][128], Kb [b][kv][s][128], Vt [b][kv][d][s],
// Out f32 [b][s][h*128+d].
// ---------------------------------------------------------------------------
__global__ __launch_bounds__(32) void attn_wmma_kernel(
    const bf16_t* __restrict__ Qb, const bf16_t* __restrict__ Kb,
    const bf16_t* __restrict__ Vt, float* __restrict__ Out) {
  __shared__ bf16_t Plds[16 * 32];

  const int qb = blockIdx.x;        // 0..127
  const int h  = blockIdx.y;        // 0..15
  const int b  = blockIdx.z;        // 0..1
  const int kv = h >> 1;
  const int lane = threadIdx.x;
  const int half = lane >> 4;
  const int l16  = lane & 15;
  const int q0   = qb * 16;
  const float scale = 0.08838834764831845f;   // 1/sqrt(128)

  const bf16_t* Qbase = Qb + (((size_t)(b * NHEAD + h)) * SS + q0) * HD;
  const bf16_t* Kbase = Kb + ((size_t)(b * NKV + kv)) * SS * HD;
  const bf16_t* Vbase = Vt + ((size_t)(b * NKV + kv)) * HD * SS;

  // Q fragments: 4 chunks of K(=d) 32 each
  v16bf aq[4];
#pragma unroll
  for (int dc = 0; dc < 4; ++dc)
    aq[dc] = load_frag_a(Qbase + (size_t)l16 * HD + dc * 32 + half * 8);

  v8f o[8];
#pragma unroll
  for (int nc = 0; nc < 8; ++nc) o[nc] = (v8f){};
  float m_i[8], l_i[8];
#pragma unroll
  for (int r = 0; r < 8; ++r) { m_i[r] = -1e30f; l_i[r] = 0.0f; }

  for (int kb = 0; kb < SS; kb += 32) {
    // --- scores: two 16x16 tiles over this 32-key block ---
    v8f c0 = (v8f){}, c1 = (v8f){};
#pragma unroll
    for (int dc = 0; dc < 4; ++dc) {
      v16bf bk0 = load_frag_b(Kbase + (size_t)(kb + l16) * HD + dc * 32 + half * 16);
      v16bf bk1 = load_frag_b(Kbase + (size_t)(kb + 16 + l16) * HD + dc * 32 + half * 16);
      c0 = wmma_bf16(aq[dc], bk0, c0);
      c1 = wmma_bf16(aq[dc], bk1, c1);
    }

    // --- online softmax update (row reductions within 16-lane halves) ---
    float p0[8], p1[8], corr[8];
#pragma unroll
    for (int r = 0; r < 8; ++r) {
      float s0 = c0[r] * scale;
      float s1 = c1[r] * scale;
      float rm = fmaxf(s0, s1);
      rm = fmaxf(rm, __shfl_xor(rm, 1, 32));
      rm = fmaxf(rm, __shfl_xor(rm, 2, 32));
      rm = fmaxf(rm, __shfl_xor(rm, 4, 32));
      rm = fmaxf(rm, __shfl_xor(rm, 8, 32));
      float mn = fmaxf(m_i[r], rm);
      corr[r] = __expf(m_i[r] - mn);
      float e0 = __expf(s0 - mn);
      float e1 = __expf(s1 - mn);
      float rs = e0 + e1;
      rs += __shfl_xor(rs, 1, 32);
      rs += __shfl_xor(rs, 2, 32);
      rs += __shfl_xor(rs, 4, 32);
      rs += __shfl_xor(rs, 8, 32);
      l_i[r] = l_i[r] * corr[r] + rs;
      m_i[r] = mn;
      p0[r] = e0;
      p1[r] = e1;
    }
#pragma unroll
    for (int nc = 0; nc < 8; ++nc)
#pragma unroll
      for (int r = 0; r < 8; ++r) o[nc][r] *= corr[r];

    // --- reshape P (C-layout) -> A-layout via LDS ---
    __syncthreads();   // protect against previous iteration's reads
    int rowoff = half * 8;
#pragma unroll
    for (int r = 0; r < 8; ++r) {
      Plds[(r + rowoff) * 32 + l16]      = (bf16_t)p0[r];
      Plds[(r + rowoff) * 32 + 16 + l16] = (bf16_t)p1[r];
    }
    __syncthreads();
    v16bf ap = load_frag_a(&Plds[l16 * 32 + half * 8]);

    // --- O += P * V ---
#pragma unroll
    for (int nc = 0; nc < 8; ++nc) {
      v16bf bv = load_frag_b(Vbase + (size_t)(nc * 16 + l16) * SS + kb + half * 16);
      o[nc] = wmma_bf16(ap, bv, o[nc]);
    }
  }

  // --- normalize and store ---
  float inv[8];
#pragma unroll
  for (int r = 0; r < 8; ++r) inv[r] = 1.0f / l_i[r];
#pragma unroll
  for (int nc = 0; nc < 8; ++nc) {
    int col = h * HD + nc * 16 + l16;
#pragma unroll
    for (int r = 0; r < 8; ++r) {
      int row = q0 + r + half * 8;
      Out[((size_t)(b * SS + row)) * HIDD + col] = o[nc][r] * inv[r];
    }
  }
}

// ---------------------------------------------------------------------------
// Host launcher
// ---------------------------------------------------------------------------
extern "C" void kernel_launch(void* const* d_in, const int* in_sizes, int n_in,
                              void* d_out, int out_size, void* d_ws, size_t ws_size,
                              hipStream_t stream) {
  const float* x    = (const float*)d_in[0];
  const float* Wq   = (const float*)d_in[1];
  const float* Wk   = (const float*)d_in[2];
  const float* Wv   = (const float*)d_in[3];
  const float* Wo   = (const float*)d_in[4];
  const float* cosb = (const float*)d_in[5];
  const float* sinb = (const float*)d_in[6];
  (void)in_sizes; (void)n_in; (void)out_size; (void)ws_size;  // start_pos==0

  char* w = (char*)d_ws;
  size_t off = 0;
  auto take = [&](size_t bytes) {
    size_t o = off;
    off += (bytes + 255) & ~(size_t)255;
    return o;
  };
  bf16_t* xb  = (bf16_t*)(w + take((size_t)MROWS * HIDD * 2));        // x bf16 (later: attn bf16)
  bf16_t* Wqt = (bf16_t*)(w + take((size_t)HIDD * HIDD * 2));
  bf16_t* Wkt = (bf16_t*)(w + take((size_t)(NKV * HD) * HIDD * 2));
  bf16_t* Wvt = (bf16_t*)(w + take((size_t)(NKV * HD) * HIDD * 2));
  bf16_t* Wot = (bf16_t*)(w + take((size_t)HIDD * HIDD * 2));
  float*  xq  = (float*)(w + take((size_t)MROWS * HIDD * 4));         // later: attn f32
  float*  xk  = (float*)(w + take((size_t)MROWS * NKV * HD * 4));
  float*  xv  = (float*)(w + take((size_t)MROWS * NKV * HD * 4));
  bf16_t* Qb  = (bf16_t*)(w + take((size_t)BB * NHEAD * SS * HD * 2));
  bf16_t* Kb  = (bf16_t*)(w + take((size_t)BB * NKV * SS * HD * 2));
  bf16_t* Vt  = (bf16_t*)(w + take((size_t)BB * NKV * HD * SS * 2));

  const int TB = 256;
  // 1. x -> bf16
  {
    int n = MROWS * HIDD;
    f32_to_bf16_kernel<<<(n + TB - 1) / TB, TB, 0, stream>>>(x, xb, n);
  }
  // 2. transpose weights -> bf16 [N][K]
  {
    int n = HIDD * HIDD;
    transpose_to_bf16_kernel<<<(n + TB - 1) / TB, TB, 0, stream>>>(Wq, Wqt, HIDD, HIDD);
    transpose_to_bf16_kernel<<<(n + TB - 1) / TB, TB, 0, stream>>>(Wo, Wot, HIDD, HIDD);
    int nk = HIDD * NKV * HD;
    transpose_to_bf16_kernel<<<(nk + TB - 1) / TB, TB, 0, stream>>>(Wk, Wkt, HIDD, NKV * HD);
    transpose_to_bf16_kernel<<<(nk + TB - 1) / TB, TB, 0, stream>>>(Wv, Wvt, HIDD, NKV * HD);
  }
  // 3. projections (WMMA GEMM)
  {
    dim3 gq(HIDD / 128, MROWS / 64);
    gemm_bf16_wmma_kernel<<<gq, 128, 0, stream>>>(xb, Wqt, xq, MROWS, HIDD, HIDD);
    dim3 gkv((NKV * HD) / 128, MROWS / 64);
    gemm_bf16_wmma_kernel<<<gkv, 128, 0, stream>>>(xb, Wkt, xk, MROWS, NKV * HD, HIDD);
    gemm_bf16_wmma_kernel<<<gkv, 128, 0, stream>>>(xb, Wvt, xv, MROWS, NKV * HD, HIDD);
  }
  // 4. RoPE + repack Q/K, transpose V
  {
    int nq = BB * SS * NHEAD * 64;
    rope_q_kernel<<<(nq + TB - 1) / TB, TB, 0, stream>>>(xq, cosb, sinb, Qb);
    int nk = BB * SS * NKV * 64;
    rope_k_kernel<<<(nk + TB - 1) / TB, TB, 0, stream>>>(xk, cosb, sinb, Kb);
    int nv = BB * NKV * HD * SS;
    v_transpose_kernel<<<(nv + TB - 1) / TB, TB, 0, stream>>>(xv, Vt);
  }
  // 5. attention (flash-style, WMMA); output f32 reuses xq buffer
  float* attn_f32 = xq;
  {
    dim3 g(SS / 16, NHEAD, BB);
    attn_wmma_kernel<<<g, 32, 0, stream>>>(Qb, Kb, Vt, attn_f32);
  }
  // 6. attn -> bf16 (reuse xb), final output GEMM into d_out (f32)
  {
    int n = MROWS * HIDD;
    f32_to_bf16_kernel<<<(n + TB - 1) / TB, TB, 0, stream>>>(attn_f32, xb, n);
    dim3 g(HIDD / 128, MROWS / 64);
    gemm_bf16_wmma_kernel<<<g, 128, 0, stream>>>(xb, Wot, (float*)d_out, MROWS, HIDD, HIDD);
  }
}
